// WindowMultiHeadAttention_5669356833262
// MI455X (gfx1250) — matrix operand order repaired and checked by reading, not compiled, not run on hardware
//
#include <hip/hip_runtime.h>
#include <hip/hip_bf16.h>

typedef unsigned short u16;
typedef __attribute__((ext_vector_type(16))) __bf16 v16bf;
typedef __attribute__((ext_vector_type(8)))  float  v8f;

__device__ __forceinline__ u16 f2bf(float f) {
    union { float f; unsigned int u; } x; x.f = f;
    unsigned int u = x.u + 0x7FFFu + ((x.u >> 16) & 1u);
    return (u16)(u >> 16);
}
__device__ __forceinline__ float bf2f(u16 h) {
    union { unsigned int u; float f; } x; x.u = ((unsigned int)h) << 16;
    return x.f;
}

// A fragment: 16x32 bf16 tile, row-major source with leading dim `ld` (elements).
// Lane L<16: row=row0+L, K-chunks [kb,kb+8) and [kb+16,kb+24) with kb=0;
// Lane L>=16: same rows, kb=8.  (ISA 7.12.2, 16-bit A-matrix layout)
__device__ __forceinline__ v16bf load_a16(const u16* base, int ld, int row, int kbase) {
    union { v16bf v; uint4 q[2]; } u;
    const u16* p = base + row * ld + kbase;
    u.q[0] = *(const uint4*)(p);
    u.q[1] = *(const uint4*)(p + 16);
    return u.v;
}
// B fragment: 32x16 bf16 tile from row-major [K][N]; p = &B[k0][n], n = L%16;
// lane half h supplies K = 8h..8h+7 and 16+8h..16+8h+7 (strided by ld).
__device__ __forceinline__ v16bf load_b16(const u16* p, int ld) {
    union { v16bf v; u16 e[16]; } u;
    const int h = (threadIdx.x >> 4) & 1;
#pragma unroll
    for (int s = 0; s < 8; ++s) {
        u.e[s]     = p[(8 * h + s) * ld];
        u.e[s + 8] = p[(16 + 8 * h + s) * ld];
    }
    return u.v;
}

__device__ __forceinline__ v8f wmma_bf16(v16bf a, v16bf b, v8f c) {
    return __builtin_amdgcn_wmma_f32_16x16x32_bf16(false, a, false, b, (short)0, c, false, false);
}

// ---------------- prelude kernels ----------------

// MLP relative position bias table: bias[h][q][k], h<16, q,k<64
__global__ void bias_table_kernel(const float* __restrict__ w_m1, const float* __restrict__ b_m1,
                                  const float* __restrict__ w_m2, const float* __restrict__ b_m2,
                                  float* __restrict__ bias_out) {
    int p = blockIdx.x * blockDim.x + threadIdx.x;
    if (p >= 4096) return;
    int a = p >> 6, c = p & 63;
    float dy = (float)((a >> 3) - (c >> 3));
    float dx = (float)((a & 7) - (c & 7));
    float ry = (dy > 0.f ? 1.f : (dy < 0.f ? -1.f : 0.f)) * log1pf(fabsf(dy));
    float rx = (dx > 0.f ? 1.f : (dx < 0.f ? -1.f : 0.f)) * log1pf(fabsf(dx));
    float acc[16];
#pragma unroll
    for (int h = 0; h < 16; ++h) acc[h] = b_m2[h];
    for (int m = 0; m < 256; ++m) {
        float hv = ry * w_m1[m * 2] + rx * w_m1[m * 2 + 1] + b_m1[m];
        hv = hv > 0.f ? hv : 0.f;
#pragma unroll
        for (int h = 0; h < 16; ++h) acc[h] += hv * w_m2[h * 256 + m];
    }
#pragma unroll
    for (int h = 0; h < 16; ++h) bias_out[h * 4096 + p] = acc[h];
}

// w [rows][cols] f32  ->  outT [cols][rows] bf16
__global__ void transpose_bf16_kernel(const float* __restrict__ w, u16* __restrict__ outT,
                                      int rows, int cols) {
    int i = blockIdx.x * blockDim.x + threadIdx.x;
    if (i >= rows * cols) return;
    int r = i / cols, c = i - r * cols;
    outT[(size_t)c * rows + r] = f2bf(w[i]);
}

// ---------------- fused window attention ----------------
// LDS layout (dynamic, 300032 B):
//   X    : [64][520] bf16                      @ 0       (66560)
//   qkv  : per-wave {q[64][32], kT[32][64], v[64][32]}  @ 66560  (8*12288)
//   attn : per-wave [64][64] bf16              @ 164864 (8*8192)
//   norms: per-wave {qn[64], kn[64]} f32       @ 230400 (8*512)
//   OUT  : [64][512] bf16                      @ 234496 (65536)
#define LD_X 520

__global__ __launch_bounds__(256, 1)
void swin_attn_kernel(const float* __restrict__ x, const float* __restrict__ mask,
                      const u16* __restrict__ wqkvT, const float* __restrict__ b_qkv,
                      const u16* __restrict__ wprojT, const float* __restrict__ b_proj,
                      const float* __restrict__ bias_tbl, const float* __restrict__ tau,
                      float* __restrict__ out) {
    extern __shared__ char smem[];
    const int b    = blockIdx.x;
    const int tid  = threadIdx.x;
    const int wv   = tid >> 5;
    const int lane = tid & 31;
    const int hi   = lane >> 4;     // lane-half selector
    const int l15  = lane & 15;

    u16*   X     = (u16*)smem;
    u16*   qbuf  = (u16*)(smem + 66560 + wv * 12288 * 2 / 2);   // [64][32]
    u16*   kTbuf = qbuf + 64 * 32;                               // [32][64]
    u16*   vbuf  = kTbuf + 32 * 64;                              // [64][32]
    u16*   attnb = (u16*)(smem + 164864 + wv * 8192);            // [64][64]
    float* norms = (float*)(smem + 230400 + wv * 512);           // qn[64], kn[64]
    u16*   OUT   = (u16*)(smem + 234496);                        // [64][512]

    // ---- phase 1: stage x[b] -> bf16 token-major panel ----
    const float* xb = x + (size_t)b * 32768;
    for (int i = tid; i < 32768; i += 256) {
        int c = i >> 6, t = i & 63;
        X[t * LD_X + c] = f2bf(xb[i]);
    }
    __syncthreads();

    const int w64 = b & 63;   // window index within mask group (nw = 64)

    for (int hh = 0; hh < 2; ++hh) {
        const int h = wv * 2 + hh;

        // ---- QKV GEMM for head h: [64 tok] x [512 C] x [96 cols] ----
#pragma unroll
        for (int nt = 0; nt < 6; ++nt) {
            const int sec = nt >> 1;                       // 0=q 1=k 2=v
            const int ncb = (nt & 1) * 16;
            const int n0  = sec * 512 + h * 32 + ncb;      // column in W_qkvT [512][1536]
            v8f acc0 = {0,0,0,0,0,0,0,0}, acc1 = {0,0,0,0,0,0,0,0};
            v8f acc2 = {0,0,0,0,0,0,0,0}, acc3 = {0,0,0,0,0,0,0,0};
#pragma unroll 4
            for (int k0 = 0; k0 < 512; k0 += 32) {
                v16bf Bt = load_b16(wqkvT + (size_t)k0 * 1536 + n0 + l15, 1536);
                const int kb = k0 + hi * 8;
                acc0 = wmma_bf16(load_a16(X, LD_X,  0 + l15, kb), Bt, acc0);
                acc1 = wmma_bf16(load_a16(X, LD_X, 16 + l15, kb), Bt, acc1);
                acc2 = wmma_bf16(load_a16(X, LD_X, 32 + l15, kb), Bt, acc2);
                acc3 = wmma_bf16(load_a16(X, LD_X, 48 + l15, kb), Bt, acc3);
            }
            const float bq = b_qkv[n0 + l15];
            v8f accs[4] = {acc0, acc1, acc2, acc3};
#pragma unroll
            for (int mt = 0; mt < 4; ++mt)
#pragma unroll
                for (int r = 0; r < 8; ++r) {
                    int row = mt * 16 + r + 8 * hi;        // token
                    u16 hv = f2bf(accs[mt][r] + bq);
                    if (sec == 0)       qbuf[row * 32 + ncb + l15] = hv;
                    else if (sec == 1)  kTbuf[(ncb + l15) * 64 + row] = hv;   // transposed
                    else                vbuf[row * 32 + ncb + l15] = hv;
                }
        }

        // ---- cosine norms ----
        for (int t = lane; t < 64; t += 32) {
            float sq = 0.f, sk = 0.f;
#pragma unroll
            for (int d = 0; d < 32; ++d) {
                float qv = bf2f(qbuf[t * 32 + d]);  sq += qv * qv;
                float kv = bf2f(kTbuf[d * 64 + t]); sk += kv * kv;
            }
            norms[t]      = sqrtf(sq);
            norms[64 + t] = sqrtf(sk);
        }

        // ---- scores: q @ k^T (K = 32 = one WMMA step) ----
        float tc = tau[h]; tc = tc < 0.01f ? 0.01f : tc;
        const float itau = 1.0f / tc;
        v8f S[4][4];
#pragma unroll
        for (int mt = 0; mt < 4; ++mt)
#pragma unroll
            for (int nt = 0; nt < 4; ++nt) { v8f z = {0,0,0,0,0,0,0,0}; S[mt][nt] = z; }
#pragma unroll
        for (int nt = 0; nt < 4; ++nt) {
            v16bf Bt = load_b16(kTbuf + nt * 16 + l15, 64);   // K=dim, N=tokens
#pragma unroll
            for (int mt = 0; mt < 4; ++mt) {
                v16bf A = load_a16(qbuf, 32, mt * 16 + l15, hi * 8);
                S[mt][nt] = wmma_bf16(A, Bt, S[mt][nt]);
            }
        }

        // ---- scale + bias + mask + softmax (register-resident, lane layout of C/D) ----
        const float* biasH = bias_tbl + (size_t)h * 4096;
        const float* maskW = mask + (size_t)w64 * 4096;
#pragma unroll
        for (int mt = 0; mt < 4; ++mt) {
#pragma unroll
            for (int r = 0; r < 8; ++r) {
                int row = mt * 16 + r + 8 * hi;
                float qn = norms[row];
                float mx = -1e30f;
#pragma unroll
                for (int nt = 0; nt < 4; ++nt) {
                    int col = nt * 16 + l15;
                    float dn = qn * norms[64 + col];
                    dn = dn < 1e-6f ? 1e-6f : dn;
                    float e = S[mt][nt][r] / dn * itau + biasH[row * 64 + col] + maskW[row * 64 + col];
                    S[mt][nt][r] = e;
                    mx = fmaxf(mx, e);
                }
#pragma unroll
                for (int s = 1; s < 16; s <<= 1) mx = fmaxf(mx, __shfl_xor(mx, s, 32));
                float sum = 0.f;
#pragma unroll
                for (int nt = 0; nt < 4; ++nt) {
                    float e = __expf(S[mt][nt][r] - mx);
                    S[mt][nt][r] = e;
                    sum += e;
                }
#pragma unroll
                for (int s = 1; s < 16; s <<= 1) sum += __shfl_xor(sum, s, 32);
                float inv = 1.0f / sum;
#pragma unroll
                for (int nt = 0; nt < 4; ++nt)
                    attnb[row * 64 + nt * 16 + l15] = f2bf(S[mt][nt][r] * inv);
            }
        }

        // ---- out = attn @ v : [64][64] x [64][32] ----
        v8f O[4][2];
#pragma unroll
        for (int mt = 0; mt < 4; ++mt)
#pragma unroll
            for (int nt = 0; nt < 2; ++nt) { v8f z = {0,0,0,0,0,0,0,0}; O[mt][nt] = z; }
#pragma unroll
        for (int kk = 0; kk < 64; kk += 32) {
#pragma unroll
            for (int nt = 0; nt < 2; ++nt) {
                v16bf Bt = load_b16(vbuf + kk * 32 + nt * 16 + l15, 32);
#pragma unroll
                for (int mt = 0; mt < 4; ++mt) {
                    v16bf A = load_a16(attnb, 64, mt * 16 + l15, kk + hi * 8);
                    O[mt][nt] = wmma_bf16(A, Bt, O[mt][nt]);
                }
            }
        }
#pragma unroll
        for (int mt = 0; mt < 4; ++mt)
#pragma unroll
            for (int nt = 0; nt < 2; ++nt)
#pragma unroll
                for (int r = 0; r < 8; ++r) {
                    int row = mt * 16 + r + 8 * hi;
                    OUT[row * 512 + h * 32 + nt * 16 + l15] = f2bf(O[mt][nt][r]);
                }
    }
    __syncthreads();

    // ---- proj: [64][512] x [512][512], wave wv owns N tiles [wv*4, wv*4+4) ----
    float* ob = out + (size_t)b * 32768;
#pragma unroll
    for (int nn = 0; nn < 4; ++nn) {
        const int n0 = (wv * 4 + nn) * 16;
        v8f P0 = {0,0,0,0,0,0,0,0}, P1 = {0,0,0,0,0,0,0,0};
        v8f P2 = {0,0,0,0,0,0,0,0}, P3 = {0,0,0,0,0,0,0,0};
#pragma unroll 4
        for (int k0 = 0; k0 < 512; k0 += 32) {
            v16bf Bt = load_b16(wprojT + (size_t)k0 * 512 + n0 + l15, 512);
            const int kb = k0 + hi * 8;
            P0 = wmma_bf16(load_a16(OUT, 512,  0 + l15, kb), Bt, P0);
            P1 = wmma_bf16(load_a16(OUT, 512, 16 + l15, kb), Bt, P1);
            P2 = wmma_bf16(load_a16(OUT, 512, 32 + l15, kb), Bt, P2);
            P3 = wmma_bf16(load_a16(OUT, 512, 48 + l15, kb), Bt, P3);
        }
        const int col = n0 + l15;               // output channel
        const float bp = b_proj[col];
        v8f Ps[4] = {P0, P1, P2, P3};
#pragma unroll
        for (int mt = 0; mt < 4; ++mt)
#pragma unroll
            for (int r = 0; r < 8; ++r) {
                int row = mt * 16 + r + 8 * hi; // token
                ob[(size_t)col * 64 + row] = Ps[mt][r] + bp;
            }
    }
}

// ---------------- launch ----------------
extern "C" void kernel_launch(void* const* d_in, const int* in_sizes, int n_in,
                              void* d_out, int out_size, void* d_ws, size_t ws_size,
                              hipStream_t stream) {
    const float* x      = (const float*)d_in[0];
    const float* mask   = (const float*)d_in[1];
    const float* w_qkv  = (const float*)d_in[2];
    const float* b_qkv  = (const float*)d_in[3];
    const float* w_proj = (const float*)d_in[4];
    const float* b_proj = (const float*)d_in[5];
    const float* w_m1   = (const float*)d_in[6];
    const float* b_m1   = (const float*)d_in[7];
    const float* w_m2   = (const float*)d_in[8];
    const float* b_m2   = (const float*)d_in[9];
    const float* tau    = (const float*)d_in[10];
    float* out = (float*)d_out;

    char* ws = (char*)d_ws;
    float* bias_tbl = (float*)ws;                         //  16*4096*4   = 262144
    u16*   wqkvT    = (u16*)(ws + 262144);                //  512*1536*2  = 1572864
    u16*   wprojT   = (u16*)(ws + 262144 + 1572864);      //  512*512*2   = 524288

    const int B = in_sizes[0] >> 15;                      // elements / (512*64)

    bias_table_kernel<<<16, 256, 0, stream>>>(w_m1, b_m1, w_m2, b_m2, bias_tbl);
    transpose_bf16_kernel<<<(1536 * 512 + 255) / 256, 256, 0, stream>>>(w_qkv, wqkvT, 1536, 512);
    transpose_bf16_kernel<<<(512 * 512 + 255) / 256, 256, 0, stream>>>(w_proj, wprojT, 512, 512);
    swin_attn_kernel<<<B, 256, 300032, stream>>>(x, mask, wqkvT, b_qkv, wprojT, b_proj,
                                                 bias_tbl, tau, out);
}
